// Block_27848567948000
// MI455X (gfx1250) — matrix operand order repaired and checked by reading, not compiled, not run on hardware
//
#include <hip/hip_runtime.h>

#define BDIM 4
#define TDIM 2048
#define CDIM 1280
#define HDIM 10
#define DDIM 128
#define FDIM 5120
#define EPSF 1e-5f

typedef __attribute__((ext_vector_type(16))) __bf16 v16bf;
typedef __attribute__((ext_vector_type(8)))  float  v8f;
typedef __attribute__((ext_vector_type(4)))  int    v4i;

union FragA { v16bf v; uint4 u[2]; };
union U128  { uint4 u; unsigned short s[8]; };

#define AS_GLOBAL __attribute__((address_space(1)))
#define AS_LOCAL  __attribute__((address_space(3)))

#if __has_builtin(__builtin_amdgcn_global_load_async_to_lds_b128)
#define USE_ASYNC_LDS 1
#else
#define USE_ASYNC_LDS 0
#endif

__device__ __forceinline__ void async_copy_b128(const unsigned short* g,
                                                unsigned short* l) {
#if USE_ASYNC_LDS
  __builtin_amdgcn_global_load_async_to_lds_b128(
      (AS_GLOBAL v4i*)g, (AS_LOCAL v4i*)l, 0, 0);
#else
  *reinterpret_cast<uint4*>(l) = *reinterpret_cast<const uint4*>(g);
#endif
}

__device__ __forceinline__ void wait_async0() {
#if USE_ASYNC_LDS
#if __has_builtin(__builtin_amdgcn_s_wait_asynccnt)
  __builtin_amdgcn_s_wait_asynccnt(0);
#else
  asm volatile("s_wait_asynccnt 0x0" ::: "memory");
#endif
#endif
}

__device__ __forceinline__ unsigned short f2bf(float f) {
  unsigned u = __float_as_uint(f);
  u += 0x7FFFu + ((u >> 16) & 1u);           // round-to-nearest-even
  return (unsigned short)(u >> 16);
}

__device__ __forceinline__ v8f wmma_bf16(const FragA& a, const FragA& b, v8f c) {
  return __builtin_amdgcn_wmma_f32_16x16x32_bf16(false, a.v, false, b.v,
                                                 (short)0, c, false, false);
}

// ---- MXFP8 (e4m3 value, UE8M0 per-32-block scale) quantize->dequantize ----
__device__ __forceinline__ float quant_e4m3(float x) {
  float ax = fabsf(x);
  int ex = (int)((__float_as_uint(ax) >> 23) & 0xFFu) - 127;
  ex = ex < -6 ? -6 : ex;                     // denorm floor: steps of 2^-9
  float step = ldexpf(1.0f, ex - 3);          // 3 mantissa bits
  float q = rintf(ax / step) * step;          // RNE
  q = fminf(q, 448.0f);
  return copysignf(q, x);
}

__device__ __forceinline__ float mx_qdq(float x, float amax) {
  amax = fmaxf(amax, 1e-12f);
  int e = (int)((__float_as_uint(amax) >> 23) & 0xFFu) - 127; // floor(log2(amax))
  int sh = e + 119;                             // +127-8
  sh = sh < 0 ? 0 : (sh > 255 ? 255 : sh);
  float xs = fminf(fmaxf(x * ldexpf(1.0f, 127 - sh), -448.0f), 448.0f);
  return quant_e4m3(xs) * ldexpf(1.0f, sh - 127);
}

// ---------------- elementwise f32 -> bf16 ----------------
__global__ void cvt_f32_bf16(const float* __restrict__ in,
                             unsigned short* __restrict__ out, long long n) {
  long long i = (long long)blockIdx.x * blockDim.x + threadIdx.x;
  long long stride = (long long)gridDim.x * blockDim.x;
  for (; i < n; i += stride) out[i] = f2bf(in[i]);
}

// ---------------- row RMSNorm -> bf16 ----------------
__global__ __launch_bounds__(256)
void rmsnorm_to_bf16(const float* __restrict__ x, const float* __restrict__ w,
                     unsigned short* __restrict__ out, int C) {
  const int row = blockIdx.x;
  const float* xr = x + (size_t)row * C;
  unsigned short* orow = out + (size_t)row * C;
  float ss = 0.f;
  for (int i = threadIdx.x; i < C; i += 256) { float v = xr[i]; ss += v * v; }
  #pragma unroll
  for (int mm = 16; mm; mm >>= 1) ss += __shfl_xor(ss, mm, 32);
  __shared__ float red[8];
  __shared__ float sinv;
  const int lane = threadIdx.x & 31, wv = threadIdx.x >> 5;
  if (lane == 0) red[wv] = ss;
  __syncthreads();
  if (threadIdx.x == 0) {
    float t = 0.f;
    for (int i = 0; i < 8; ++i) t += red[i];
    sinv = rsqrtf(t / (float)C + EPSF);
  }
  __syncthreads();
  const float inv = sinv;
  for (int i = threadIdx.x; i < C; i += 256)
    orow[i] = f2bf(xr[i] * inv * w[i]);
}

// ---------------- bf16 WMMA GEMM: C(MxN) = A(MxK) @ B(KxN) ----------------
// EPI: 0 = f32 out, 1 = f32 out + residual, 2 = bf16 out of srelu(acc)
template<int EPI>
__global__ __launch_bounds__(256)
void gemm_bf16(const unsigned short* __restrict__ A,
               const unsigned short* __restrict__ B,
               const float* __restrict__ resid,
               float* __restrict__ outF,
               unsigned short* __restrict__ outB,
               int M, int N, int K) {
  __shared__ unsigned short sA[64 * 32];   // [m][k], row stride 32
  __shared__ unsigned short sBT[64 * 32];  // [n][k], row stride 32
  const int tid = threadIdx.x;
  const int lane = tid & 31, wv = tid >> 5;
  const int wm = wv >> 2, wn = wv & 3;
  const int bm = blockIdx.y * 64, bn = blockIdx.x * 64;
  const int cl = lane & 15, hfl = lane >> 4;

  v8f acc0 = {0,0,0,0,0,0,0,0};
  v8f acc1 = {0,0,0,0,0,0,0,0};

  const int arow = tid >> 2, acol = (tid & 3) * 8;   // A: 64 rows x 32 cols
  const int bk = tid >> 3,  bn8 = (tid & 7) * 8;     // B: 32 rows x 64 cols

  for (int k0 = 0; k0 < K; k0 += 32) {
    // A tile: straight copy -> async global->LDS (ASYNCcnt path)
    async_copy_b128(&A[(size_t)(bm + arow) * K + k0 + acol], &sA[arow * 32 + acol]);
    // B tile: transposed stage (manual; async loads cannot transpose)
    U128 bw; bw.u = *reinterpret_cast<const uint4*>(&B[(size_t)(k0 + bk) * N + bn + bn8]);
    #pragma unroll
    for (int j = 0; j < 8; ++j) sBT[(bn8 + j) * 32 + bk] = bw.s[j];
    if (k0 + 32 < K) {
      __builtin_prefetch(&A[(size_t)(bm + arow) * K + k0 + 32 + acol], 0, 3);
      __builtin_prefetch(&B[(size_t)(k0 + 32 + bk) * N + bn + bn8], 0, 3);
    }
    wait_async0();
    __syncthreads();

    FragA bf;
    const int nn = wn * 16 + cl;
    bf.u[0] = *reinterpret_cast<const uint4*>(&sBT[nn * 32 + hfl * 8]);
    bf.u[1] = *reinterpret_cast<const uint4*>(&sBT[nn * 32 + 16 + hfl * 8]);
    FragA a0, a1;
    const int r0 = wm * 32 + cl, r1 = r0 + 16;
    a0.u[0] = *reinterpret_cast<const uint4*>(&sA[r0 * 32 + hfl * 8]);
    a0.u[1] = *reinterpret_cast<const uint4*>(&sA[r0 * 32 + 16 + hfl * 8]);
    a1.u[0] = *reinterpret_cast<const uint4*>(&sA[r1 * 32 + hfl * 8]);
    a1.u[1] = *reinterpret_cast<const uint4*>(&sA[r1 * 32 + 16 + hfl * 8]);
    acc0 = wmma_bf16(a0, bf, acc0);
    acc1 = wmma_bf16(a1, bf, acc1);
    __syncthreads();
  }

  const int nc = bn + wn * 16 + cl;
  #pragma unroll
  for (int r = 0; r < 8; ++r) {
    const int m0 = bm + wm * 32 + r + 8 * hfl;
    const int m1 = m0 + 16;
    const size_t i0 = (size_t)m0 * N + nc, i1 = (size_t)m1 * N + nc;
    if (EPI == 0) {
      outF[i0] = acc0[r]; outF[i1] = acc1[r];
    } else if (EPI == 1) {
      outF[i0] = acc0[r] + resid[i0]; outF[i1] = acc1[r] + resid[i1];
    } else {
      float t0 = fmaxf(acc0[r], 0.f), t1 = fmaxf(acc1[r], 0.f);
      outB[i0] = f2bf(t0 * t0); outB[i1] = f2bf(t1 * t1);
    }
  }
}

// ------- RoPE + per-head RMSNorm (q,k) + MXFP8 qdq (q,k,v) -> bf16 [B][H][T][D] -------
__global__ __launch_bounds__(320)
void rope_norm_mxfp8(const float* __restrict__ qkv, const float* __restrict__ freqs,
                     const float* __restrict__ qnw, const float* __restrict__ knw,
                     unsigned short* __restrict__ qd, unsigned short* __restrict__ kd,
                     unsigned short* __restrict__ vd) {
  const int t = blockIdx.x, b = blockIdx.y;
  const int h = threadIdx.x >> 5, lane = threadIdx.x & 31;  // one wave per head
  const size_t ro = ((size_t)(b * TDIM + t)) * (3 * CDIM) + (size_t)h * DDIM;
  const int d0 = 2 * lane, d1 = d0 + 1;
  const float* fr = freqs + (size_t)t * DDIM;
  const float c0 = cosf(fr[d0]),      sn0 = sinf(fr[d0]);
  const float c1 = cosf(fr[d1]),      sn1 = sinf(fr[d1]);
  const float c2 = cosf(fr[d0 + 64]), sn2 = sinf(fr[d0 + 64]);
  const float c3 = cosf(fr[d1 + 64]), sn3 = sinf(fr[d1 + 64]);

  float q0 = qkv[ro + d0], q1 = qkv[ro + d1];
  float q2 = qkv[ro + d0 + 64], q3 = qkv[ro + d1 + 64];
  float k0 = qkv[ro + CDIM + d0], k1 = qkv[ro + CDIM + d1];
  float k2 = qkv[ro + CDIM + d0 + 64], k3 = qkv[ro + CDIM + d1 + 64];
  float v0 = qkv[ro + 2 * CDIM + d0], v1 = qkv[ro + 2 * CDIM + d1];
  float v2 = qkv[ro + 2 * CDIM + d0 + 64], v3 = qkv[ro + 2 * CDIM + d1 + 64];

  // rotate-half RoPE
  float a0 = q0 * c0 - q2 * sn0, a1 = q1 * c1 - q3 * sn1;
  float a2 = q2 * c2 + q0 * sn2, a3 = q3 * c3 + q1 * sn3;
  float e0 = k0 * c0 - k2 * sn0, e1 = k1 * c1 - k3 * sn1;
  float e2 = k2 * c2 + k0 * sn2, e3 = k3 * c3 + k1 * sn3;

  // per-head RMSNorm (D=128 across the wave)
  float ssq = a0 * a0 + a1 * a1 + a2 * a2 + a3 * a3;
  float ssk = e0 * e0 + e1 * e1 + e2 * e2 + e3 * e3;
  #pragma unroll
  for (int mm = 16; mm; mm >>= 1) {
    ssq += __shfl_xor(ssq, mm, 32);
    ssk += __shfl_xor(ssk, mm, 32);
  }
  const float qi = rsqrtf(ssq * (1.0f / DDIM) + EPSF);
  const float ki = rsqrtf(ssk * (1.0f / DDIM) + EPSF);
  a0 *= qi * qnw[d0]; a1 *= qi * qnw[d1]; a2 *= qi * qnw[d0 + 64]; a3 *= qi * qnw[d1 + 64];
  e0 *= ki * knw[d0]; e1 *= ki * knw[d1]; e2 *= ki * knw[d0 + 64]; e3 *= ki * knw[d1 + 64];

  // MXFP8 qdq: 32-element blocks == 16-lane halves of the wave
  auto blk = [&](float& x, float& y) {
    float am = fmaxf(fabsf(x), fabsf(y));
    #pragma unroll
    for (int mm = 8; mm; mm >>= 1) am = fmaxf(am, __shfl_xor(am, mm, 32));
    x = mx_qdq(x, am);
    y = mx_qdq(y, am);
  };
  blk(a0, a1); blk(a2, a3);
  blk(e0, e1); blk(e2, e3);
  blk(v0, v1); blk(v2, v3);

  const size_t ob = (((size_t)(b * HDIM + h)) * TDIM + t) * DDIM;
  qd[ob + d0] = f2bf(a0); qd[ob + d1] = f2bf(a1);
  qd[ob + d0 + 64] = f2bf(a2); qd[ob + d1 + 64] = f2bf(a3);
  kd[ob + d0] = f2bf(e0); kd[ob + d1] = f2bf(e1);
  kd[ob + d0 + 64] = f2bf(e2); kd[ob + d1 + 64] = f2bf(e3);
  vd[ob + d0] = f2bf(v0); vd[ob + d1] = f2bf(v1);
  vd[ob + d0 + 64] = f2bf(v2); vd[ob + d1 + 64] = f2bf(v3);
}

// ---------------- causal flash attention, bf16 WMMA ----------------
// Q,K,V: bf16 [B][H][T][D].  O: bf16 [B][T][H][D].
__global__ __launch_bounds__(256)
void attn_causal(const unsigned short* __restrict__ Qd,
                 const unsigned short* __restrict__ Kd,
                 const unsigned short* __restrict__ Vd,
                 unsigned short* __restrict__ O) {
  __shared__ unsigned short sK[32 * 128];   // [kcol][d]
  __shared__ unsigned short sVT[128 * 32];  // [d][kcol]
  __shared__ unsigned short sP[8 * 16 * 32];// per-wave 16x32 P staging
  const int qb = blockIdx.x;
  const int bh = blockIdx.y;                 // b*H + h
  const int b = bh / HDIM, h = bh % HDIM;
  const size_t base = (size_t)bh * TDIM * DDIM;
  const int tid = threadIdx.x, lane = tid & 31, wv = tid >> 5;
  const int cl = lane & 15, hfl = lane >> 4;

  // Load this wave's Q fragments (16 rows x 128)
  const int qrow = qb * 128 + wv * 16 + cl;
  FragA aq[4];
  #pragma unroll
  for (int kk = 0; kk < 4; ++kk) {
    const size_t qo = base + (size_t)qrow * DDIM + kk * 32 + hfl * 16;
    aq[kk].u[0] = *reinterpret_cast<const uint4*>(&Qd[qo]);
    aq[kk].u[1] = *reinterpret_cast<const uint4*>(&Qd[qo + 8]);
  }

  v8f accO[8];
  #pragma unroll
  for (int i = 0; i < 8; ++i) accO[i] = (v8f){0,0,0,0,0,0,0,0};
  float m8[8], l8[8];
  #pragma unroll
  for (int r = 0; r < 8; ++r) { m8[r] = -3.0e38f; l8[r] = 0.f; }

  const int wmaxrow = qb * 128 + wv * 16 + 15;
  const int lr = tid >> 3, seg = (tid & 7) * 16;
  const int nkt = 4 * (qb + 1);

  for (int kt = 0; kt < nkt; ++kt) {
    // stage K (32x128) via async global->LDS, V^T (128x32) manually
    {
      const size_t kro = base + (size_t)(kt * 32 + lr) * DDIM + seg;
      async_copy_b128(&Kd[kro], &sK[lr * 128 + seg]);
      async_copy_b128(&Kd[kro + 8], &sK[lr * 128 + seg + 8]);
      U128 vv0, vv1;
      vv0.u = *reinterpret_cast<const uint4*>(&Vd[kro]);
      vv1.u = *reinterpret_cast<const uint4*>(&Vd[kro + 8]);
      #pragma unroll
      for (int j = 0; j < 8; ++j) {
        sVT[(seg + j) * 32 + lr] = vv0.s[j];
        sVT[(seg + 8 + j) * 32 + lr] = vv1.s[j];
      }
    }
    wait_async0();
    __syncthreads();
    const bool active = (kt * 32) <= wmaxrow;
    if (active) {
      v8f sc0 = (v8f){0,0,0,0,0,0,0,0};
      v8f sc1 = (v8f){0,0,0,0,0,0,0,0};
      #pragma unroll
      for (int kk = 0; kk < 4; ++kk) {
        FragA bk0, bk1;
        bk0.u[0] = *reinterpret_cast<const uint4*>(&sK[cl * 128 + kk * 32 + hfl * 16]);
        bk0.u[1] = *reinterpret_cast<const uint4*>(&sK[cl * 128 + kk * 32 + hfl * 16 + 8]);
        bk1.u[0] = *reinterpret_cast<const uint4*>(&sK[(16 + cl) * 128 + kk * 32 + hfl * 16]);
        bk1.u[1] = *reinterpret_cast<const uint4*>(&sK[(16 + cl) * 128 + kk * 32 + hfl * 16 + 8]);
        sc0 = wmma_bf16(aq[kk], bk0, sc0);
        sc1 = wmma_bf16(aq[kk], bk1, sc1);
      }
      const float scale = 0.088388347648318447f; // 1/sqrt(128)
      float p0[8], p1[8], mn[8];
      const int kc0 = kt * 32 + cl, kc1 = kc0 + 16;
      #pragma unroll
      for (int r = 0; r < 8; ++r) {
        const int qg = qb * 128 + wv * 16 + r + 8 * hfl;
        float x0 = sc0[r] * scale, x1 = sc1[r] * scale;
        if (kc0 > qg) x0 = -3.0e38f;
        if (kc1 > qg) x1 = -3.0e38f;
        p0[r] = x0; p1[r] = x1;
        mn[r] = fmaxf(x0, x1);
      }
      #pragma unroll
      for (int mm = 8; mm; mm >>= 1)
        #pragma unroll
        for (int r = 0; r < 8; ++r) mn[r] = fmaxf(mn[r], __shfl_xor(mn[r], mm, 32));
      float alpha[8];
      #pragma unroll
      for (int r = 0; r < 8; ++r) {
        float mnew = fmaxf(m8[r], mn[r]);
        alpha[r] = __expf(m8[r] - mnew);
        m8[r] = mnew;
        p0[r] = __expf(p0[r] - mnew);
        p1[r] = __expf(p1[r] - mnew);
      }
      float rs[8];
      #pragma unroll
      for (int r = 0; r < 8; ++r) rs[r] = p0[r] + p1[r];
      #pragma unroll
      for (int mm = 8; mm; mm >>= 1)
        #pragma unroll
        for (int r = 0; r < 8; ++r) rs[r] += __shfl_xor(rs[r], mm, 32);
      #pragma unroll
      for (int r = 0; r < 8; ++r) l8[r] = l8[r] * alpha[r] + rs[r];
      #pragma unroll
      for (int i = 0; i < 8; ++i)
        #pragma unroll
        for (int r = 0; r < 8; ++r) accO[i][r] *= alpha[r];
      // stage P as bf16 in per-wave LDS region
      #pragma unroll
      for (int r = 0; r < 8; ++r) {
        sP[wv * 512 + (r + 8 * hfl) * 32 + cl] = f2bf(p0[r]);
        sP[wv * 512 + (r + 8 * hfl) * 32 + 16 + cl] = f2bf(p1[r]);
      }
    }
    __syncthreads();
    if (active) {
      FragA ap;
      ap.u[0] = *reinterpret_cast<const uint4*>(&sP[wv * 512 + cl * 32 + hfl * 16]);
      ap.u[1] = *reinterpret_cast<const uint4*>(&sP[wv * 512 + cl * 32 + hfl * 16 + 8]);
      #pragma unroll
      for (int dsl = 0; dsl < 8; ++dsl) {
        FragA bv;
        bv.u[0] = *reinterpret_cast<const uint4*>(&sVT[(dsl * 16 + cl) * 32 + hfl * 16]);
        bv.u[1] = *reinterpret_cast<const uint4*>(&sVT[(dsl * 16 + cl) * 32 + hfl * 16 + 8]);
        accO[dsl] = wmma_bf16(ap, bv, accO[dsl]);
      }
    }
    __syncthreads();
  }

  #pragma unroll
  for (int dsl = 0; dsl < 8; ++dsl)
    #pragma unroll
    for (int r = 0; r < 8; ++r) {
      const int tq = qb * 128 + wv * 16 + r + 8 * hfl;
      const size_t oidx = (((size_t)b * TDIM + tq) * HDIM + h) * DDIM + dsl * 16 + cl;
      O[oidx] = f2bf(accO[dsl][r] / l8[r]);
    }
}

extern "C" void kernel_launch(void* const* d_in, const int* in_sizes, int n_in,
                              void* d_out, int out_size, void* d_ws, size_t ws_size,
                              hipStream_t stream) {
  const float* x    = (const float*)d_in[0];
  const float* rope = (const float*)d_in[1];
  const float* ln1  = (const float*)d_in[2];
  const float* wqkv = (const float*)d_in[3];
  const float* qnw  = (const float*)d_in[4];
  const float* knw  = (const float*)d_in[5];
  const float* wout = (const float*)d_in[6];
  const float* ln2  = (const float*)d_in[7];
  const float* wfc1 = (const float*)d_in[8];
  const float* wfc2 = (const float*)d_in[9];
  float* out = (float*)d_out;

  const size_t BT = (size_t)BDIM * TDIM;  // 8192
  char* wsb = (char*)d_ws;
  size_t off = 0;
  auto take = [&](size_t bytes) {
    char* p = wsb + off;
    off += (bytes + 255) & ~(size_t)255;
    return p;
  };
  unsigned short* xn    = (unsigned short*)take(BT * CDIM * 2);
  unsigned short* wqkvb = (unsigned short*)take((size_t)CDIM * 3 * CDIM * 2);
  unsigned short* woutb = (unsigned short*)take((size_t)CDIM * CDIM * 2);
  unsigned short* wfc1b = (unsigned short*)take((size_t)CDIM * FDIM * 2);
  unsigned short* wfc2b = (unsigned short*)take((size_t)FDIM * CDIM * 2);
  float*          qkvf  = (float*)take(BT * 3 * CDIM * 4);
  unsigned short* qd    = (unsigned short*)take(BT * CDIM * 2);
  unsigned short* kd    = (unsigned short*)take(BT * CDIM * 2);
  unsigned short* vd    = (unsigned short*)take(BT * CDIM * 2);
  unsigned short* obuf  = (unsigned short*)take(BT * CDIM * 2);
  float*          x2    = (float*)take(BT * CDIM * 4);
  unsigned short* hn    = (unsigned short*)take(BT * CDIM * 2);
  unsigned short* hbuf  = (unsigned short*)take(BT * FDIM * 2);

  // weights -> bf16 (done every call; deterministic)
  cvt_f32_bf16<<<2048, 256, 0, stream>>>(wqkv, wqkvb, (long long)CDIM * 3 * CDIM);
  cvt_f32_bf16<<<2048, 256, 0, stream>>>(wout, woutb, (long long)CDIM * CDIM);
  cvt_f32_bf16<<<2048, 256, 0, stream>>>(wfc1, wfc1b, (long long)CDIM * FDIM);
  cvt_f32_bf16<<<2048, 256, 0, stream>>>(wfc2, wfc2b, (long long)FDIM * CDIM);

  // 1) ln1 RMSNorm
  rmsnorm_to_bf16<<<(int)BT, 256, 0, stream>>>(x, ln1, xn, CDIM);
  // 2) QKV GEMM (8192 x 3840 x 1280)
  gemm_bf16<0><<<dim3(3 * CDIM / 64, (int)BT / 64), 256, 0, stream>>>(
      xn, wqkvb, nullptr, qkvf, nullptr, (int)BT, 3 * CDIM, CDIM);
  // 3) RoPE + q/k head RMSNorm + MXFP8 qdq
  rope_norm_mxfp8<<<dim3(TDIM, BDIM), 320, 0, stream>>>(qkvf, rope, qnw, knw, qd, kd, vd);
  // 4) causal attention
  attn_causal<<<dim3(TDIM / 128, BDIM * HDIM), 256, 0, stream>>>(qd, kd, vd, obuf);
  // 5) out-proj + residual
  gemm_bf16<1><<<dim3(CDIM / 64, (int)BT / 64), 256, 0, stream>>>(
      obuf, woutb, x, x2, nullptr, (int)BT, CDIM, CDIM);
  // 6) ln2 RMSNorm
  rmsnorm_to_bf16<<<(int)BT, 256, 0, stream>>>(x2, ln2, hn, CDIM);
  // 7) FC1 + squared ReLU
  gemm_bf16<2><<<dim3(FDIM / 64, (int)BT / 64), 256, 0, stream>>>(
      hn, wfc1b, nullptr, nullptr, hbuf, (int)BT, FDIM, CDIM);
  // 8) FC2 + residual -> output
  gemm_bf16<1><<<dim3(CDIM / 64, (int)BT / 64), 256, 0, stream>>>(
      hbuf, wfc2b, x2, out, nullptr, (int)BT, CDIM, FDIM);

  (void)in_sizes; (void)n_in; (void)out_size; (void)ws_size;
}
